// ObjectCondensation_30932354466029
// MI455X (gfx1250) — compile-verified, matrix operand order checked by reading, not run on hardware
//
#include <hip/hip_runtime.h>
#include <math.h>

#define NK 1024
#define NB 4
#define CD 8
#define Q_MIN_C 0.1f
#define BETA_EPS_C 0.001f
#define LSE_EPS_C 0.001f

typedef __attribute__((ext_vector_type(16))) _Float16 v16h;
typedef __attribute__((ext_vector_type(8)))  float    v8f;
typedef __attribute__((ext_vector_type(4)))  float    v4f;
typedef __attribute__((ext_vector_type(4)))  unsigned int v4u;

union V16HU { v16h v; _Float16 h[16]; unsigned int u[8]; };

// batch accumulator slots (32 floats total)
#define BA_ATT  0
#define BA_REP  4
#define BA_PEN  8
#define BA_CNT  12
#define BA_NCNT 16
#define BA_NSUM 20

__device__ __forceinline__ float qfun(float b) {
    float bc = fminf(fmaxf(b, 0.f), 1.f - BETA_EPS_C);
    float a  = atanhf(bc);
    return a * a + Q_MIN_C;
}

__device__ __forceinline__ int compute_rs(int n, const long long* splits) {
    int rs = 0;
#pragma unroll
    for (int j = 1; j < NB; ++j) rs += ((long long)n >= splits[j]) ? 1 : 0;
    return rs;
}

// ---------------- init ----------------
__global__ void k_init(unsigned int* segbits, float* Nk, int* alpha,
                       float* lse, float* sumb, float* psum, float* batch) {
    int i = blockIdx.x * blockDim.x + threadIdx.x;
    if (i < NK) {
        segbits[i] = 0u; Nk[i] = 0.f; alpha[i] = -1;
        lse[i] = 0.f; sumb[i] = 0.f; psum[i] = 0.f;
    }
    if (i < 32) batch[i] = 0.f;
}

// ---------------- pass 1: seg_max (bits trick, beta>0) + N_k ----------------
__global__ void k_p1(const float* beta, const long long* asso,
                     unsigned int* segbits, float* Nk, int n_pts) {
    int n = blockIdx.x * blockDim.x + threadIdx.x;
    if (n >= n_pts) return;
    int k = (int)asso[n];
    if (k >= 0) {
        atomicMax(&segbits[k], __float_as_uint(beta[n]));
        atomicAdd(&Nk[k], 1.f);
    }
}

// ---------------- pass 2: alpha = argmax (max index among ties) ----------------
__global__ void k_p2(const float* beta, const long long* asso,
                     const unsigned int* segbits, int* alpha, int n_pts) {
    int n = blockIdx.x * blockDim.x + threadIdx.x;
    if (n >= n_pts) return;
    int k = (int)asso[n];
    if (k >= 0 && __float_as_uint(beta[n]) == segbits[k]) atomicMax(&alpha[k], n);
}

// ---------------- per-cluster gather ----------------
__global__ void k_cluster(const float* beta, const float* coords, const long long* splits,
                          const float* Nk, const int* alpha,
                          float* ck2, float* wk, float* qalpha, int* objrs,
                          unsigned int* xah) {
    int k = blockIdx.x * blockDim.x + threadIdx.x;
    if (k >= NK) return;
    int a = alpha[k];
    bool objv = (a >= 0);
    int ai = objv ? a : 0;
    float s = 0.f;
    _Float16 hh[CD];
#pragma unroll
    for (int c = 0; c < CD; ++c) {
        float x = coords[(size_t)ai * CD + c];
        s += x * x;
        hh[c] = (_Float16)x;
    }
    ck2[k] = s;
    float qa = objv ? qfun(beta[ai]) : 0.f;
    qalpha[k] = qa;
    wk[k] = qa / fmaxf(Nk[k], 1.f);       // == 0 when cluster invalid
    objrs[k] = objv ? compute_rs(ai, splits) : -1;
    union { _Float16 h[2]; unsigned int u; } cv;
#pragma unroll
    for (int t = 0; t < 4; ++t) {
        cv.h[0] = hh[2 * t]; cv.h[1] = hh[2 * t + 1];
        xah[k * 4 + t] = cv.u;
    }
}

// ---------------- pass 3: attractive term + per-cluster/batch reductions ----------------
__global__ void k_p3(const float* beta, const float* coords, const long long* asso,
                     const long long* splits, const unsigned int* segbits,
                     const int* alpha, const float* qalpha,
                     float* att, float* lse, float* sumb, float* psum,
                     float* batch, int n_pts) {
    int n = blockIdx.x * blockDim.x + threadIdx.x;
    if (n >= n_pts) return;
    float b = beta[n];
    int k = (int)asso[n];
    int rs = compute_rs(n, splits);
    if (k >= 0) {
        float q = qfun(b);
        int ai = max(alpha[k], 0);
        float d = 0.f;
#pragma unroll
        for (int c = 0; c < CD; ++c) {
            float t = coords[(size_t)n * CD + c] - coords[(size_t)ai * CD + c];
            d += t * t;
        }
        float att_n = q * qalpha[k] * d;
        att[n] = att_n;
        atomicAdd(&batch[BA_ATT + rs], att_n);
        float m = __uint_as_float(segbits[k]);
        atomicAdd(&lse[k], expf((b - m) / LSE_EPS_C));
        atomicAdd(&sumb[k], b);
        float bc = fminf(fmaxf(b, 0.f), 1.f - BETA_EPS_C);
        float a2 = atanhf(bc); a2 = a2 * a2;
        atomicAdd(&psum[k], fmaxf(a2, 1e-6f));
    } else {
        att[n] = 0.f;
        atomicAdd(&batch[BA_NCNT + rs], 1.f);
        atomicAdd(&batch[BA_NSUM + rs], b);
    }
}

// ---------------- per-cluster beta penalty ----------------
__global__ void k_cluster2(const unsigned int* segbits, const float* lse_sum,
                           const float* sumb, const int* objrs, float* batch) {
    int k = blockIdx.x * blockDim.x + threadIdx.x;
    if (k >= NK) return;
    int ors = objrs[k];
    if (ors >= 0) {
        float m = __uint_as_float(segbits[k]);
        float lse = m + LSE_EPS_C * logf(lse_sum[k] + 1e-30f);
        float pen = 1.f - lse + (1.f - fminf(fmaxf(sumb[k], 0.f), 1.f));
        atomicAdd(&batch[BA_PEN + ors], pen);
        atomicAdd(&batch[BA_CNT + ors], 1.f);
    }
}

// ---------------- repulsion: WMMA over 16pt x 16cluster tiles ----------------
__global__ __launch_bounds__(256) void k_rep(const float* coords, const float* beta,
                                             const long long* asso, const long long* splits,
                                             const unsigned int* xah, const float* wk,
                                             const float* ck2, const int* objrs,
                                             const float* att, float* out_lvrep,
                                             float* batch, int n_pts, int n_tiles) {
    // f16 x_alpha table (8 halves/cluster) + 16-cluster zero pad for lanes 16..31
    __shared__ __align__(16) unsigned int s_xa[(NK + 16) * 4];
    // packed per-cluster metadata: {w, ck2, bits(obj_rs), 0}
    __shared__ __align__(16) float s_meta[NK * 4];
    for (int i = threadIdx.x; i < NK; i += blockDim.x) {
        s_meta[4 * i + 0] = wk[i];
        s_meta[4 * i + 1] = ck2[i];
        s_meta[4 * i + 2] = __int_as_float(objrs[i]);
        s_meta[4 * i + 3] = 0.f;
    }
    for (int i = threadIdx.x; i < NK * 4; i += blockDim.x) s_xa[i] = xah[i];
    for (int i = NK * 4 + threadIdx.x; i < (NK + 16) * 4; i += blockDim.x) s_xa[i] = 0u;
    __syncthreads();

    int wid  = blockIdx.x * (blockDim.x >> 5) + (threadIdx.x >> 5);
    int lane = threadIdx.x & 31;
    if (wid >= n_tiles) return;          // wave-uniform: EXEC stays all-ones for WMMA
    int mbase = wid * 16;
    int half  = lane >> 4;

    // lanes 0..15 own one point-row each: build A (16x32 f16, K=0..7 live) + metadata
    V16HU A;
#pragma unroll
    for (int j = 0; j < 16; ++j) A.h[j] = (_Float16)0.f;
    float cn2_self = 0.f, q_self = 0.f;
    int ptrs_self = -9, am_self = -2;
    if (lane < 16) {
        int row = mbase + lane;
        int rl  = min(row, n_pts - 1);
        float s = 0.f;
#pragma unroll
        for (int c = 0; c < CD; ++c) {
            float x = coords[(size_t)rl * CD + c];
            s += x * x;
            A.h[c] = (_Float16)x;
        }
        cn2_self = s;
        q_self   = qfun(beta[rl]);
        int kk   = (int)asso[rl];
        am_self  = (kk >= 0) ? kk : -2;
        ptrs_self = (row < n_pts) ? compute_rs(rl, splits) : -9;
    }
    // broadcast the 8 rows this lane's D-slice covers
    float cn2r[8], qr[8]; int pr[8], ar[8];
#pragma unroll
    for (int r = 0; r < 8; ++r) {
        int src = half * 8 + r;
        cn2r[r] = __shfl(cn2_self, src, 32);
        qr[r]   = __shfl(q_self,  src, 32);
        pr[r]   = __shfl(ptrs_self, src, 32);
        ar[r]   = __shfl(am_self,  src, 32);
    }

    float acc[8];
#pragma unroll
    for (int r = 0; r < 8; ++r) acc[r] = 0.f;

    // branchless per-lane LDS cursors: high lanes read the zero pad (step 0)
    unsigned int ba    = (lane < 16) ? (unsigned)(lane * 4) : (unsigned)((NK + lane - 16) * 4);
    unsigned int bstep = (lane < 16) ? 64u : 0u;
    unsigned int ma    = (unsigned)((lane & 15) * 4);

    for (int kt = 0; kt < NK / 16; ++kt) {
        v4u braw = *(const v4u*)(s_xa + ba);   ba += bstep;
        v4f meta = *(const v4f*)(s_meta + ma); ma += 64u;

        V16HU Bm;
        Bm.u[0] = braw[0]; Bm.u[1] = braw[1]; Bm.u[2] = braw[2]; Bm.u[3] = braw[3];
        Bm.u[4] = 0u; Bm.u[5] = 0u; Bm.u[6] = 0u; Bm.u[7] = 0u;

        v8f D = {};
        D = __builtin_amdgcn_wmma_f32_16x16x32_f16(false, A.v, false, Bm.v,
                                                   (short)0, D, false, false);
        float wn  = meta[0];
        float c2n = meta[1];
        int   orn = __float_as_int(meta[2]);
        int  ncol = kt * 16 + (lane & 15);
#pragma unroll
        for (int r = 0; r < 8; ++r) {
            float dsq = fmaxf(fmaf(-2.f, D[r], cn2r[r] + c2n), 0.f);
            float sq  = __builtin_amdgcn_sqrtf(dsq + 1e-6f);
            float vr  = fmaxf(1.f - sq, 0.f);
            // branchless mask: obj_valid folded into wn (==0 for invalid clusters)
            int msk = (pr[r] == orn) & (ar[r] != ncol);
            float wsel = msk ? wn : 0.f;
            acc[r] = fmaf(vr, wsel, acc[r]);
        }
    }
    // butterfly reduce over the 16 lanes of each half (xor 1,2,4,8 stays in-half)
#pragma unroll
    for (int r = 0; r < 8; ++r) {
        float v = acc[r];
        v += __shfl_xor(v, 1, 32);
        v += __shfl_xor(v, 2, 32);
        v += __shfl_xor(v, 4, 32);
        v += __shfl_xor(v, 8, 32);
        acc[r] = v;
    }
    if ((lane & 15) == 0) {
        int rb = mbase + half * 8;
#pragma unroll
        for (int r = 0; r < 8; ++r) {
            int m = rb + r;
            if (m < n_pts) {
                float rep = qr[r] * acc[r];
                out_lvrep[m] = att[m] + rep;
                atomicAdd(&batch[BA_REP + pr[r]], rep);
            }
        }
    }
}

// ---------------- pl ----------------
__global__ void k_pl(const float* beta, const long long* asso, const long long* splits,
                     const float* psum, const float* batch, float* out_pl, int n_pts) {
    int n = blockIdx.x * blockDim.x + threadIdx.x;
    if (n >= n_pts) return;
    int k = (int)asso[n];
    int rs = compute_rs(n, splits);
    float pl = 0.f;
    if (k >= 0) {
        float bc = fminf(fmaxf(beta[n], 0.f), 1.f - BETA_EPS_C);
        float a = atanhf(bc);
        float p = fmaxf(a * a, 1e-6f);
        pl = p / fmaxf(psum[k], 1e-6f);
    }
    pl = pl / fmaxf(batch[BA_CNT + rs], 1.f) / (float)NB;
    out_pl[n] = pl;
}

// ---------------- finalize scalars ----------------
__global__ void k_fin(const long long* splits, const float* batch, float* out) {
    if (blockIdx.x == 0 && threadIdx.x == 0) {
        float lv = 0.f, lr = 0.f, lb = 0.f;
#pragma unroll
        for (int b = 0; b < NB; ++b) {
            float len = (float)(splits[b + 1] - splits[b]);
            lv += batch[BA_ATT + b] / len;
            lr += batch[BA_REP + b] / len;
            float pen = batch[BA_PEN + b] / fmaxf(batch[BA_CNT + b], 1.f);
            float nm  = batch[BA_NSUM + b] / fmaxf(batch[BA_NCNT + b], 1.f);
            lb += pen + nm;  // S_B = 1.0
        }
        out[0] = lv / (float)NB;
        out[1] = lr / (float)NB;
        out[2] = lb / (float)NB;
    }
}

extern "C" void kernel_launch(void* const* d_in, const int* in_sizes, int n_in,
                              void* d_out, int out_size, void* d_ws, size_t ws_size,
                              hipStream_t stream) {
    const float*     beta   = (const float*)d_in[0];
    const float*     coords = (const float*)d_in[1];
    const long long* asso   = (const long long*)d_in[2];
    const long long* splits = (const long long*)d_in[3];
    int n_pts = in_sizes[0];

    float* ws = (float*)d_ws;
    float*        att     = ws;
    unsigned int* segbits = (unsigned int*)(ws + n_pts);
    float*        Nk      = ws + n_pts + 1 * NK;
    int*          alpha   = (int*)(ws + n_pts + 2 * NK);
    float*        ck2     = ws + n_pts + 3 * NK;
    float*        wk      = ws + n_pts + 4 * NK;
    float*        qalpha  = ws + n_pts + 5 * NK;
    int*          objrs   = (int*)(ws + n_pts + 6 * NK);
    float*        lse     = ws + n_pts + 7 * NK;
    float*        sumb    = ws + n_pts + 8 * NK;
    float*        psum    = ws + n_pts + 9 * NK;
    unsigned int* xah     = (unsigned int*)(ws + n_pts + 10 * NK);  // NK*4 words
    float*        batch   = ws + n_pts + 14 * NK;                   // 32 floats

    float* out       = (float*)d_out;
    float* out_pl    = out + 3;
    float* out_lvrep = out + 3 + n_pts;

    const int TPB = 256;
    int nb_pts = (n_pts + TPB - 1) / TPB;
    int nb_k   = (NK + TPB - 1) / TPB;

    k_init<<<nb_k, TPB, 0, stream>>>(segbits, Nk, alpha, lse, sumb, psum, batch);
    k_p1<<<nb_pts, TPB, 0, stream>>>(beta, asso, segbits, Nk, n_pts);
    k_p2<<<nb_pts, TPB, 0, stream>>>(beta, asso, segbits, alpha, n_pts);
    k_cluster<<<nb_k, TPB, 0, stream>>>(beta, coords, splits, Nk, alpha,
                                        ck2, wk, qalpha, objrs, xah);
    k_p3<<<nb_pts, TPB, 0, stream>>>(beta, coords, asso, splits, segbits, alpha,
                                     qalpha, att, lse, sumb, psum, batch, n_pts);
    k_cluster2<<<nb_k, TPB, 0, stream>>>(segbits, lse, sumb, objrs, batch);

    int n_tiles = (n_pts + 15) / 16;
    int waves_per_block = TPB / 32;
    int nb_rep = (n_tiles + waves_per_block - 1) / waves_per_block;
    k_rep<<<nb_rep, TPB, 0, stream>>>(coords, beta, asso, splits, xah, wk, ck2,
                                      objrs, att, out_lvrep, batch, n_pts, n_tiles);

    k_pl<<<nb_pts, TPB, 0, stream>>>(beta, asso, splits, psum, batch, out_pl, n_pts);
    k_fin<<<1, 64, 0, stream>>>(splits, batch, out);
}
